// EncodeCoreDecode_84619445666528
// MI455X (gfx1250) — matrix operand order repaired and checked by reading, not compiled, not run on hardware
//
#include <hip/hip_runtime.h>
#include <hip/hip_bf16.h>

typedef __attribute__((ext_vector_type(16))) __bf16    v16bf;
typedef __attribute__((ext_vector_type(8)))  __bf16    v8bf;
typedef __attribute__((ext_vector_type(2)))  __bf16    v2bf;
typedef __attribute__((ext_vector_type(8)))  float     v8f;
typedef __attribute__((ext_vector_type(8)))  unsigned  v8u;

#define N_EDGES 200000
#define N_NODES 20000
#define N_GRAPHS 64
#define OUT_STEP (N_EDGES + N_NODES + N_GRAPHS)
#define LN_EPS 1e-5f

// pack two f32 -> dword holding {bf16(f1), bf16(f0)}, round-to-nearest-even
__device__ __forceinline__ unsigned pack2(float f0, float f1) {
#if defined(__has_builtin) && __has_builtin(__builtin_amdgcn_cvt_pk_bf16_f32)
  v2bf p = __builtin_amdgcn_cvt_pk_bf16_f32(f0, f1);
  return __builtin_bit_cast(unsigned, p);
#else
  unsigned a = __builtin_bit_cast(unsigned, f0);
  a += 0x7FFFu + ((a >> 16) & 1u);
  unsigned b = __builtin_bit_cast(unsigned, f1);
  b += 0x7FFFu + ((b >> 16) & 1u);
  return (a >> 16) | (b & 0xFFFF0000u);
#endif
}
__device__ __forceinline__ v16bf pack16(float4 a, float4 b, float4 c, float4 d) {
  v8u u;
  u[0] = pack2(a.x, a.y); u[1] = pack2(a.z, a.w);
  u[2] = pack2(b.x, b.y); u[3] = pack2(b.z, b.w);
  u[4] = pack2(c.x, c.y); u[5] = pack2(c.z, c.w);
  u[6] = pack2(d.x, d.y); u[7] = pack2(d.z, d.w);
  return __builtin_bit_cast(v16bf, u);
}

// ---------------------------------------------------------------------------
// Tiled bf16-WMMA GEMM: out[M,128] = A[M,K] @ W[K,128]  (+ fused epilogue)
//   HAS_ADD0 : add add0[row, :]
//   NGATHER  : add gN[idxN[row], :] for N=1..NGATHER
//   DO_LN    : + bias, LayerNorm(gamma,beta), ReLU
//   DO_OUT   : after LN/ReLU, dot with outw[128] + outb -> out[row] (scalar)
// Block: 256 threads = 8 waves; wave w computes rows [16w,16w+16) x 128 cols.
// W panel is pre-swizzled into exact B-fragment layout in LDS (b128 accesses);
// A fragments are loaded directly global->VGPR (each lane owns one row).
// ---------------------------------------------------------------------------
template<int K, int NGATHER, bool HAS_ADD0, bool DO_LN, bool DO_OUT>
__launch_bounds__(256)
__global__ void gemm_bf16_kernel(
    const float* __restrict__ A, int M,
    const float* __restrict__ W,
    const float* __restrict__ add0,
    const float* __restrict__ g1buf, const int* __restrict__ g1idx,
    const float* __restrict__ g2buf, const int* __restrict__ g2idx,
    const float* __restrict__ g3buf, const int* __restrict__ g3idx,
    const float* __restrict__ bias,
    const float* __restrict__ gamma, const float* __restrict__ beta,
    float* __restrict__ out,
    const float* __restrict__ outw, const float* __restrict__ outb)
{
  static_assert(K % 32 == 0, "K must be multiple of 32");
  // B-fragment-layout LDS: [K/32 chunks][8 nblk][2 seg][32 lanes][8 bf16]
  __shared__ __bf16 Wswz[(size_t)K * 128];

  const int tid  = threadIdx.x;
  const int wv   = tid >> 5;
  const int lane = tid & 31;
  const int half = lane >> 4;
  const int l16  = lane & 15;
  const int rowBase = blockIdx.x * 128;

  // ---- Stage W -> LDS in B-fragment layout (16B vector stores) ----
  for (int gi = tid; gi < K * 16; gi += 256) {
    int lane_s = gi & 31;
    int seg    = (gi >> 5) & 1;
    int b      = (gi >> 6) & 7;
    int chunk  = gi >> 9;
    int col    = b * 16 + (lane_s & 15);
    int row0   = chunk * 32 + (lane_s >> 4) * 8 + seg * 16;
    const float* wp = W + (size_t)row0 * 128 + col;
    uint4 st;
    st.x = pack2(wp[0],   wp[128]);
    st.y = pack2(wp[256], wp[384]);
    st.z = pack2(wp[512], wp[640]);
    st.w = pack2(wp[768], wp[896]);
    *(uint4*)(&Wswz[(size_t)gi * 8]) = st;
  }
  __syncthreads();

  v8f acc[8];
#pragma unroll
  for (int b = 0; b < 8; ++b)
#pragma unroll
    for (int i = 0; i < 8; ++i) acc[b][i] = 0.f;

  // This lane's A row (fragment rows are per-lane: m = 16*wv + l16)
  const int gm_row = rowBase + wv * 16 + l16;
  const bool rowOK = (gm_row < M);
  const float* Arow = A + (size_t)gm_row * K;

  for (int kc = 0; kc < K; kc += 32) {
    // A fragment: lane needs cols {k0..k0+7, k0+16..k0+23}, k0 = 8*half
    float4 fa0, fa1, fa2, fa3;
    if (rowOK) {
      const float4* p = (const float4*)(Arow + kc + half * 8);
      fa0 = p[0]; fa1 = p[1];
      const float4* q = (const float4*)(Arow + kc + half * 8 + 16);
      fa2 = q[0]; fa3 = q[1];
    } else {
      fa0 = fa1 = fa2 = fa3 = make_float4(0.f, 0.f, 0.f, 0.f);
    }
    v16bf af = pack16(fa0, fa1, fa2, fa3);

    const int cbase = ((kc >> 5) * 8) * 512;   // chunk base (elems)
#pragma unroll
    for (int b = 0; b < 8; ++b) {
      const __bf16* bp = &Wswz[cbase + b * 512 + lane * 8];
      v8bf lo = *(const v8bf*)bp;            // seg 0: elems 0..7
      v8bf hi = *(const v8bf*)(bp + 256);    // seg 1: elems 8..15
      v16bf bfrag = __builtin_shufflevector(lo, hi,
          0, 1, 2, 3, 4, 5, 6, 7, 8, 9, 10, 11, 12, 13, 14, 15);
      acc[b] = __builtin_amdgcn_wmma_f32_16x16x32_bf16(
          false, af, false, bfrag, (short)0, acc[b], false, false);
    }
  }

  // ---------------- Epilogue ----------------
  // Thread holds 8 rows (r) x 8 col-blocks (b): row = rbase+r, col = 16b+l16
  const int rbase = rowBase + wv * 16 + half * 8;
  float v[8][8];
#pragma unroll
  for (int b = 0; b < 8; ++b)
#pragma unroll
    for (int r = 0; r < 8; ++r) v[b][r] = acc[b][r];

  if (HAS_ADD0) {
#pragma unroll
    for (int r = 0; r < 8; ++r) {
      int gr = rbase + r;
      if (gr < M) {
#pragma unroll
        for (int b = 0; b < 8; ++b)
          v[b][r] += add0[(size_t)gr * 128 + b * 16 + l16];
      }
    }
  }
  if (NGATHER >= 1) {
#pragma unroll
    for (int r = 0; r < 8; ++r) {
      int gr = rbase + r;
      if (gr < M) {
        size_t gi = (size_t)g1idx[gr];
#pragma unroll
        for (int b = 0; b < 8; ++b) v[b][r] += g1buf[gi * 128 + b * 16 + l16];
      }
    }
  }
  if (NGATHER >= 2) {
#pragma unroll
    for (int r = 0; r < 8; ++r) {
      int gr = rbase + r;
      if (gr < M) {
        size_t gi = (size_t)g2idx[gr];
#pragma unroll
        for (int b = 0; b < 8; ++b) v[b][r] += g2buf[gi * 128 + b * 16 + l16];
      }
    }
  }
  if (NGATHER >= 3) {
#pragma unroll
    for (int r = 0; r < 8; ++r) {
      int gr = rbase + r;
      if (gr < M) {
        size_t gi = (size_t)g3idx[gr];
#pragma unroll
        for (int b = 0; b < 8; ++b) v[b][r] += g3buf[gi * 128 + b * 16 + l16];
      }
    }
  }

  if (DO_LN) {
#pragma unroll
    for (int b = 0; b < 8; ++b) {
      float bb = bias[b * 16 + l16];
#pragma unroll
      for (int r = 0; r < 8; ++r) v[b][r] += bb;
    }
    float mean[8], rstd[8];
#pragma unroll
    for (int r = 0; r < 8; ++r) {
      float s1 = 0.f, s2 = 0.f;
#pragma unroll
      for (int b = 0; b < 8; ++b) { s1 += v[b][r]; s2 += v[b][r] * v[b][r]; }
#pragma unroll
      for (int msk = 8; msk >= 1; msk >>= 1) {
        s1 += __shfl_xor(s1, msk, 32);
        s2 += __shfl_xor(s2, msk, 32);
      }
      float mu  = s1 * (1.f / 128.f);
      float var = s2 * (1.f / 128.f) - mu * mu;
      mean[r] = mu;
      rstd[r] = rsqrtf(var + LN_EPS);
    }
#pragma unroll
    for (int b = 0; b < 8; ++b) {
      float gmv = gamma[b * 16 + l16];
      float btv = beta[b * 16 + l16];
#pragma unroll
      for (int r = 0; r < 8; ++r)
        v[b][r] = fmaxf((v[b][r] - mean[r]) * rstd[r] * gmv + btv, 0.f);
    }
  }

  if (DO_OUT) {
    float ob = outb[0];
    float ow[8];
#pragma unroll
    for (int b = 0; b < 8; ++b) ow[b] = outw[b * 16 + l16];
#pragma unroll
    for (int r = 0; r < 8; ++r) {
      float d = 0.f;
#pragma unroll
      for (int b = 0; b < 8; ++b) d += v[b][r] * ow[b];
#pragma unroll
      for (int msk = 8; msk >= 1; msk >>= 1) d += __shfl_xor(d, msk, 32);
      int gr = rbase + r;
      if (l16 == 0 && gr < M) out[gr] = d + ob;
    }
  } else {
#pragma unroll
    for (int r = 0; r < 8; ++r) {
      int gr = rbase + r;
      if (gr < M) {
#pragma unroll
        for (int b = 0; b < 8; ++b)
          out[(size_t)gr * 128 + b * 16 + l16] = v[b][r];
      }
    }
  }
}

// ---------------------------------------------------------------------------
// Small helpers
// ---------------------------------------------------------------------------
__global__ void fill_zero_kernel(float* p, long n) {
  long i = (long)blockIdx.x * blockDim.x + threadIdx.x;
  if (i < n) p[i] = 0.f;
}

// dst[idx[r], c] += src[r, c]  (128 cols) via hardware f32 atomics
__global__ void scatter_rows_atomic(const float* __restrict__ src,
                                    const int* __restrict__ idx,
                                    float* __restrict__ dst, int rows) {
  long t = (long)blockIdx.x * 256 + threadIdx.x;
  if (t >= (long)rows * 128) return;
  int r = (int)(t >> 7), c = (int)(t & 127);
  unsafeAtomicAdd(&dst[(size_t)idx[r] * 128 + c], src[t]);
}

// Graph-level segment_sum (64 segments, 128 cols): LDS-staged to tame contention
__global__ void scatter_graph_lds(const float* __restrict__ src,
                                  const int* __restrict__ idx,
                                  float* __restrict__ dst, int rows, int chunk) {
  __shared__ float accum[N_GRAPHS * 128];
  for (int i = threadIdx.x; i < N_GRAPHS * 128; i += 256) accum[i] = 0.f;
  __syncthreads();
  long base = (long)blockIdx.x * chunk * 128;
  long lim  = (long)rows * 128;
  long end  = base + (long)chunk * 128;
  if (end > lim) end = lim;
  for (long t = base + threadIdx.x; t < end; t += 256) {
    int r = (int)(t >> 7), c = (int)(t & 127);
    atomicAdd(&accum[idx[r] * 128 + c], src[t]);   // ds_add_f32
  }
  __syncthreads();
  for (int i = threadIdx.x; i < N_GRAPHS * 128; i += 256) {
    float vv = accum[i];
    if (vv != 0.f) unsafeAtomicAdd(&dst[i], vv);
  }
}

// out[64,32] = relu(LN(in[64,IC] @ W[IC,32] + b));  one wave per row
template<int IC>
__global__ void small_dense_ln32(const float* __restrict__ in,
                                 const float* __restrict__ W,
                                 const float* __restrict__ b,
                                 const float* __restrict__ gm,
                                 const float* __restrict__ bt,
                                 float* __restrict__ out) {
  int row = blockIdx.x, c = threadIdx.x;
  float s = b[c];
  for (int i = 0; i < IC; ++i) s += in[row * IC + i] * W[i * 32 + c];
  float s1 = s, s2 = s * s;
  for (int m = 16; m >= 1; m >>= 1) { s1 += __shfl_xor(s1, m, 32); s2 += __shfl_xor(s2, m, 32); }
  float mu = s1 * (1.f / 32.f), var = s2 * (1.f / 32.f) - mu * mu;
  float r  = (s - mu) * rsqrtf(var + LN_EPS) * gm[c] + bt[c];
  out[row * 32 + c] = fmaxf(r, 0.f);
}

// core_g: gout[64,32] = relu(LN([g0|gin|nagg|eagg] @ W[320,32] + b))
__global__ void core_g_kernel(const float* __restrict__ g0,
                              const float* __restrict__ gin,
                              const float* __restrict__ nagg,
                              const float* __restrict__ eagg,
                              const float* __restrict__ W,
                              const float* __restrict__ b,
                              const float* __restrict__ gm,
                              const float* __restrict__ bt,
                              float* __restrict__ gout) {
  int row = blockIdx.x, c = threadIdx.x;
  float s = b[c];
  for (int i = 0; i < 32; ++i)  s += g0[row * 32 + i]   * W[i * 32 + c];
  for (int i = 0; i < 32; ++i)  s += gin[row * 32 + i]  * W[(32 + i) * 32 + c];
  for (int i = 0; i < 128; ++i) s += nagg[row * 128 + i] * W[(64 + i) * 32 + c];
  for (int i = 0; i < 128; ++i) s += eagg[row * 128 + i] * W[(192 + i) * 32 + c];
  float s1 = s, s2 = s * s;
  for (int m = 16; m >= 1; m >>= 1) { s1 += __shfl_xor(s1, m, 32); s2 += __shfl_xor(s2, m, 32); }
  float mu = s1 * (1.f / 32.f), var = s2 * (1.f / 32.f) - mu * mu;
  float r  = (s - mu) * rsqrtf(var + LN_EPS) * gm[c] + bt[c];
  gout[row * 32 + c] = fmaxf(r, 0.f);
}

// dec_g + out_g fused: outp[row] = relu(LN(gin@Wd + bd)) . ow + ob
__global__ void dec_out_g_kernel(const float* __restrict__ gin,
                                 const float* __restrict__ W,
                                 const float* __restrict__ b,
                                 const float* __restrict__ gm,
                                 const float* __restrict__ bt,
                                 const float* __restrict__ ow,
                                 const float* __restrict__ ob,
                                 float* __restrict__ outp) {
  int row = blockIdx.x, c = threadIdx.x;
  float s = b[c];
  for (int i = 0; i < 32; ++i) s += gin[row * 32 + i] * W[i * 32 + c];
  float s1 = s, s2 = s * s;
  for (int m = 16; m >= 1; m >>= 1) { s1 += __shfl_xor(s1, m, 32); s2 += __shfl_xor(s2, m, 32); }
  float mu = s1 * (1.f / 32.f), var = s2 * (1.f / 32.f) - mu * mu;
  float r  = fmaxf((s - mu) * rsqrtf(var + LN_EPS) * gm[c] + bt[c], 0.f);
  float d  = r * ow[c];
  for (int m = 16; m >= 1; m >>= 1) d += __shfl_xor(d, m, 32);
  if (c == 0) outp[row] = d + ob[0];
}

// Pg: out[64,128] = ga[64,32]@Wa[32,128] + gb[64,32]@Wb[32,128]
__global__ void pg_kernel(const float* __restrict__ ga, const float* __restrict__ gb,
                          const float* __restrict__ Wa, const float* __restrict__ Wb,
                          float* __restrict__ out) {
  int row = blockIdx.x, c = threadIdx.x;   // grid 64, block 128
  float s = 0.f;
  for (int i = 0; i < 32; ++i)
    s += ga[row * 32 + i] * Wa[i * 128 + c] + gb[row * 32 + i] * Wb[i * 128 + c];
  out[row * 128 + c] = s;
}

// ---------------------------------------------------------------------------
extern "C" void kernel_launch(void* const* d_in, const int* in_sizes, int n_in,
                              void* d_out, int out_size, void* d_ws, size_t ws_size,
                              hipStream_t stream) {
  (void)in_sizes; (void)n_in; (void)out_size; (void)ws_size;

  const float* x = (const float*)d_in[0];
  const float* e = (const float*)d_in[1];
  const float* g = (const float*)d_in[2];
  const int* edges    = (const int*)d_in[3];
  const int* node_idx = (const int*)d_in[4];
  const int* edge_idx = (const int*)d_in[5];
  const int* src = edges;
  const int* dst = edges + N_EDGES;

  const float* P[49];
  for (int i = 7; i < 49; ++i) P[i] = (const float*)d_in[i];
  const float *enc_e_W=P[7],  *enc_e_b=P[8],  *enc_e_g=P[9],  *enc_e_be=P[10];
  const float *enc_n_W=P[11], *enc_n_b=P[12], *enc_n_g=P[13], *enc_n_be=P[14];
  const float *enc_g_W=P[15], *enc_g_b=P[16], *enc_g_g=P[17], *enc_g_be=P[18];
  const float *core_e_W=P[19],*core_e_b=P[20],*core_e_g=P[21],*core_e_be=P[22];
  const float *core_n_W=P[23],*core_n_b=P[24],*core_n_g=P[25],*core_n_be=P[26];
  const float *core_g_W=P[27],*core_g_b=P[28],*core_g_g=P[29],*core_g_be=P[30];
  const float *dec_e_W=P[31], *dec_e_b=P[32], *dec_e_g=P[33], *dec_e_be=P[34];
  const float *dec_n_W=P[35], *dec_n_b=P[36], *dec_n_g=P[37], *dec_n_be=P[38];
  const float *dec_g_W=P[39], *dec_g_b=P[40], *dec_g_g=P[41], *dec_g_be=P[42];
  const float *out_e_W=P[43], *out_e_b=P[44];
  const float *out_n_W=P[45], *out_n_b=P[46];
  const float *out_g_W=P[47], *out_g_b=P[48];

  // core_e_W slices (row-major [832,128], row stride 128)
  const float *We_e0  = core_e_W;             const float *We_ec = core_e_W + 128*128;
  const float *We_x0s = core_e_W + 256*128;   const float *We_xcs = core_e_W + 384*128;
  const float *We_x0d = core_e_W + 512*128;   const float *We_xcd = core_e_W + 640*128;
  const float *We_g0  = core_e_W + 768*128;   const float *We_gc  = core_e_W + 800*128;
  // core_n_W slices ([448,128])
  const float *Wn_x0 = core_n_W;              const float *Wn_xc = core_n_W + 128*128;
  const float *Wn_ea = core_n_W + 256*128;
  const float *Wn_g0 = core_n_W + 384*128;    const float *Wn_gc = core_n_W + 416*128;

  // Workspace layout
  float* ws = (float*)d_ws;
  size_t off = 0;
  auto alloc = [&](size_t n) { float* p = ws + off; off += n; return p; };
  float* e0    = alloc((size_t)N_EDGES * 128);
  float* Ae0   = alloc((size_t)N_EDGES * 128);
  float* ecur  = alloc((size_t)N_EDGES * 128);
  float* x0    = alloc((size_t)N_NODES * 128);
  float* X0n   = alloc((size_t)N_NODES * 128);
  float* Bx0s  = alloc((size_t)N_NODES * 128);
  float* Bx0d  = alloc((size_t)N_NODES * 128);
  float* xcur  = alloc((size_t)N_NODES * 128);
  float* Ps    = alloc((size_t)N_NODES * 128);
  float* Pd    = alloc((size_t)N_NODES * 128);
  float* partn = alloc((size_t)N_NODES * 128);
  float* eaggn = alloc((size_t)N_NODES * 128);
  float* nagg  = alloc((size_t)N_GRAPHS * 128);
  float* eaggg = alloc((size_t)N_GRAPHS * 128);  // adjacent to nagg: zero together
  float* PgE   = alloc((size_t)N_GRAPHS * 128);
  float* PgN   = alloc((size_t)N_GRAPHS * 128);
  float* g0    = alloc((size_t)N_GRAPHS * 32);
  float* gcur  = alloc((size_t)N_GRAPHS * 32);

  float* dout = (float*)d_out;

  const int GB_E = (N_EDGES + 127) / 128;   // 1563
  const int GB_N = (N_NODES + 127) / 128;   // 157
  const float* NF = nullptr; const int* NI = nullptr;

  // ---- Encoders + step-invariant precomputes ----
  gemm_bf16_kernel<64, 0, false, true, false><<<GB_E, 256, 0, stream>>>(
      e, N_EDGES, enc_e_W, NF, NF, NI, NF, NI, NF, NI,
      enc_e_b, enc_e_g, enc_e_be, e0, NF, NF);
  gemm_bf16_kernel<128, 0, false, true, false><<<GB_N, 256, 0, stream>>>(
      x, N_NODES, enc_n_W, NF, NF, NI, NF, NI, NF, NI,
      enc_n_b, enc_n_g, enc_n_be, x0, NF, NF);
  small_dense_ln32<16><<<N_GRAPHS, 32, 0, stream>>>(g, enc_g_W, enc_g_b, enc_g_g, enc_g_be, g0);

  gemm_bf16_kernel<128, 0, false, false, false><<<GB_E, 256, 0, stream>>>(
      e0, N_EDGES, We_e0, NF, NF, NI, NF, NI, NF, NI, NF, NF, NF, Ae0, NF, NF);
  gemm_bf16_kernel<128, 0, false, false, false><<<GB_N, 256, 0, stream>>>(
      x0, N_NODES, We_x0s, NF, NF, NI, NF, NI, NF, NI, NF, NF, NF, Bx0s, NF, NF);
  gemm_bf16_kernel<128, 0, false, false, false><<<GB_N, 256, 0, stream>>>(
      x0, N_NODES, We_x0d, NF, NF, NI, NF, NI, NF, NI, NF, NF, NF, Bx0d, NF, NF);
  gemm_bf16_kernel<128, 0, false, false, false><<<GB_N, 256, 0, stream>>>(
      x0, N_NODES, Wn_x0, NF, NF, NI, NF, NI, NF, NI, NF, NF, NF, X0n, NF, NF);

  // ---- Message-passing steps (reference STEPS == 2) ----
  for (int s = 0; s < 2; ++s) {
    const float* ecur_in = (s == 0) ? e0 : ecur;
    const float* xcur_in = (s == 0) ? x0 : xcur;
    const float* gcur_in = (s == 0) ? g0 : gcur;

    // -- edge block --
    pg_kernel<<<N_GRAPHS, 128, 0, stream>>>(g0, gcur_in, We_g0, We_gc, PgE);
    gemm_bf16_kernel<128, 0, true, false, false><<<GB_N, 256, 0, stream>>>(
        xcur_in, N_NODES, We_xcs, Bx0s, NF, NI, NF, NI, NF, NI, NF, NF, NF, Ps, NF, NF);
    gemm_bf16_kernel<128, 0, true, false, false><<<GB_N, 256, 0, stream>>>(
        xcur_in, N_NODES, We_xcd, Bx0d, NF, NI, NF, NI, NF, NI, NF, NF, NF, Pd, NF, NF);
    gemm_bf16_kernel<128, 3, true, true, false><<<GB_E, 256, 0, stream>>>(
        ecur_in, N_EDGES, We_ec, Ae0, Ps, src, Pd, dst, PgE, edge_idx,
        core_e_b, core_e_g, core_e_be, ecur, NF, NF);

    // -- node block --
    fill_zero_kernel<<<((long)N_NODES * 128 + 255) / 256, 256, 0, stream>>>(eaggn, (long)N_NODES * 128);
    scatter_rows_atomic<<<((long)N_EDGES * 128 + 255) / 256, 256, 0, stream>>>(ecur, dst, eaggn, N_EDGES);
    pg_kernel<<<N_GRAPHS, 128, 0, stream>>>(g0, gcur_in, Wn_g0, Wn_gc, PgN);
    gemm_bf16_kernel<128, 0, true, false, false><<<GB_N, 256, 0, stream>>>(
        xcur_in, N_NODES, Wn_xc, X0n, NF, NI, NF, NI, NF, NI, NF, NF, NF, partn, NF, NF);
    gemm_bf16_kernel<128, 1, true, true, false><<<GB_N, 256, 0, stream>>>(
        eaggn, N_NODES, Wn_ea, partn, PgN, node_idx, NF, NI, NF, NI,
        core_n_b, core_n_g, core_n_be, xcur, NF, NF);

    // -- global block --
    fill_zero_kernel<<<(2 * N_GRAPHS * 128 + 255) / 256, 256, 0, stream>>>(nagg, 2 * N_GRAPHS * 128);
    scatter_graph_lds<<<(N_NODES + 511) / 512, 256, 0, stream>>>(xcur, node_idx, nagg, N_NODES, 512);
    scatter_graph_lds<<<(N_EDGES + 511) / 512, 256, 0, stream>>>(ecur, edge_idx, eaggg, N_EDGES, 512);
    core_g_kernel<<<N_GRAPHS, 32, 0, stream>>>(g0, gcur_in, nagg, eaggg, core_g_W,
                                               core_g_b, core_g_g, core_g_be, gcur);

    // -- decode + output transform (fused) --
    float* step_out = dout + (size_t)s * OUT_STEP;
    gemm_bf16_kernel<128, 0, false, true, true><<<GB_E, 256, 0, stream>>>(
        ecur, N_EDGES, dec_e_W, NF, NF, NI, NF, NI, NF, NI,
        dec_e_b, dec_e_g, dec_e_be, step_out, out_e_W, out_e_b);
    gemm_bf16_kernel<128, 0, false, true, true><<<GB_N, 256, 0, stream>>>(
        xcur, N_NODES, dec_n_W, NF, NF, NI, NF, NI, NF, NI,
        dec_n_b, dec_n_g, dec_n_be, step_out + N_EDGES, out_n_W, out_n_b);
    dec_out_g_kernel<<<N_GRAPHS, 32, 0, stream>>>(gcur, dec_g_W, dec_g_b, dec_g_g, dec_g_be,
                                                  out_g_W, out_g_b, step_out + N_EDGES + N_NODES);
  }
}
